// QGCN_38689065402672
// MI455X (gfx1250) — matrix operand order repaired and checked by reading, not compiled, not run on hardware
//
#include <hip/hip_runtime.h>
#include <hip/hip_fp16.h>

// ---------------------------------------------------------------------------
// QGCN forward on gfx1250 (MI455X), wave32 + v_wmma_f32_16x16x32_f16.
//
// Pipeline (all fp16 WMMA, fp32 accumulate):
//   P0: adj_h  = f16( min(adj/alpha,1) quantized to 15 levels * alpha )  [N x N]
//   P1: s1T    = (x_h @ w1)^T as f16                                     [H x N]
//   P2: h_h    = relu(adj_h @ s1 + b1) as f16 row-major                  [N x H]
//   P3: s2T    = (h_h @ w2)^T as f16, padded 40->48 cols (pad rows = 0)  [48 x N]
//   P4: logits = adj_h @ s2 + b2  -> d_out fp32 [N x 40]
//   P5: row softmax over the 40 logits (one wave per row)
//
// GEMM: K is a template constant (saddr addressing, offsets in the 24-bit
// immediate). Rotating 2-tile-lookahead pipeline (only NT+2 fragment slots
// live -> no spills) with sched_group_barrier pinning the schedule to
// [2*NT+2 VMEM reads][NT WMMAs][prefetch] per 32-wide K step, which keeps
// every WMMA's operand load >= 6 load-slots ahead (partial waits, never 0).
// ---------------------------------------------------------------------------

typedef __attribute__((ext_vector_type(16))) _Float16 v16h;
typedef __attribute__((ext_vector_type(8)))  _Float16 v8h;
typedef __attribute__((ext_vector_type(4)))  _Float16 v4h;
typedef __attribute__((ext_vector_type(8)))  float    v8f;
typedef __attribute__((ext_vector_type(4)))  float    v4f;

// ---- P0: 4-bit STE quantize of adj, stored as fp16 --------------------------
__global__ void quantize_adj_f16_kernel(const float* __restrict__ adj,
                                        const float* __restrict__ alphap,
                                        _Float16* __restrict__ out, long n4) {
  long i = (long)blockIdx.x * blockDim.x + threadIdx.x;
  if (i >= n4) return;
  float alpha = alphap[0];
  float scale = alpha * (1.0f / 15.0f);
  v4f v = ((const v4f*)adj)[i];
  v4h o;
#pragma unroll
  for (int j = 0; j < 4; ++j) {
    float xc = fminf(v[j] / alpha, 1.0f);
    o[j] = (_Float16)(rintf(xc * 15.0f) * scale);   // jnp.round == rint (RTE)
  }
  ((v4h*)out)[i] = o;
}

// ---- fp32 -> fp16 bulk convert (for x) -------------------------------------
__global__ void f32_to_f16_kernel(const float* __restrict__ in,
                                  _Float16* __restrict__ out, long n4) {
  long i = (long)blockIdx.x * blockDim.x + threadIdx.x;
  if (i >= n4) return;
  v4f v = ((const v4f*)in)[i];
  v4h o;
#pragma unroll
  for (int j = 0; j < 4; ++j) o[j] = (_Float16)v[j];
  ((v4h*)out)[i] = o;
}

// ---- transpose fp32 [K x Nreal] -> fp16 [Npad x K], pad rows zeroed --------
__global__ void transpose_to_f16_kernel(const float* __restrict__ in,
                                        _Float16* __restrict__ out,
                                        int K, int Nreal, int Npad) {
  int i = blockIdx.x * blockDim.x + threadIdx.x;
  if (i >= Npad * K) return;
  int n = i / K, k = i % K;
  out[i] = (_Float16)((n < Nreal) ? in[(size_t)k * Nreal + n] : 0.0f);
}

// ---- generic WMMA GEMM: C[M x Npad] = A[M x K] * BT[Npad x K]^T ------------
// Each wave computes a 16 x (NT*16) strip: A streamed once, NT accumulators.
// Fragment layouts follow CDNA5 ISA 7.12.2 (wave32, 16-bit A/B, f32 C/D).
template <int NT, int KC>
__global__ void __attribute__((amdgpu_waves_per_eu(1)))
wmma_gemm_kernel(const _Float16* __restrict__ A,
                 const _Float16* __restrict__ BT,
                 const float* __restrict__ bias,
                 float* __restrict__ C, int ldc,
                 _Float16* __restrict__ OutRM,   // [M x NT*16]
                 _Float16* __restrict__ OutT,    // [NT*16 x M]
                 int M, int Nreal, int relu) {
  static_assert(NT >= 2, "need >= 2 N-tiles");
  static_assert(KC % 64 == 0, "K must be a multiple of 64");
  const int Npad = NT * 16;
  const int lane = threadIdx.x & 31;
  const int wave = threadIdx.x >> 5;
  const int gw   = blockIdx.x * (blockDim.x >> 5) + wave;
  const int m0   = gw * 16;
  const int hsel = lane >> 4;   // which half of the wave
  const int l15  = lane & 15;

  // 32-bit element offsets; bases stay uniform => saddr-form global loads,
  // per-tile displacement t*16*KC folds into the 24-bit immediate.
  const unsigned arow = (unsigned)(m0 + l15) * KC + 8u * (unsigned)hsel;
  const unsigned bcol = (unsigned)l15 * KC + 8u * (unsigned)hsel;

  auto loadA = [&](int k) -> v16h {
    const _Float16* p = A + arow + (unsigned)k;
    v8h lo = *(const v8h*)(p);
    v8h hi = *(const v8h*)(p + 16);
    return __builtin_shufflevector(lo, hi, 0, 1, 2, 3, 4, 5, 6, 7,
                                   8, 9, 10, 11, 12, 13, 14, 15);
  };
  auto loadB = [&](int t, int k) -> v16h {
    const _Float16* p = BT + bcol + (unsigned)(t * 16 * KC + k);
    v8h lo = *(const v8h*)(p);
    v8h hi = *(const v8h*)(p + 16);
    return __builtin_shufflevector(lo, hi, 0, 1, 2, 3, 4, 5, 6, 7,
                                   8, 9, 10, 11, 12, 13, 14, 15);
  };

  v8f acc[NT];
#pragma unroll
  for (int t = 0; t < NT; ++t)
#pragma unroll
    for (int r = 0; r < 8; ++r) acc[t][r] = 0.0f;

  // Rotating pipeline, lookahead = 2 tiles (NT+2 fragment slots live).
  v16h a  = loadA(0);
  v16h b0 = loadB(0, 0);
  v16h b1 = loadB(1, 0);

  for (int k = 0; k < KC; k += 32) {
    const int kn = (k + 32 < KC) ? (k + 32) : 0;  // safe wrap on final step
    v16h a_n = a;
#pragma unroll
    for (int t = 0; t < NT; ++t) {
      v16h b2;
      if (t + 2 < NT) {
        b2 = loadB(t + 2, k);
      } else if (t + 2 == NT) {
        a_n = loadA(kn);
        b2  = loadB(0, kn);
      } else {  // t == NT-1
        b2 = loadB(1, kn);
      }
      acc[t] = __builtin_amdgcn_wmma_f32_16x16x32_f16(
          false, a, false, b0, (short)0, acc[t], false, false);
      b0 = b1;
      b1 = b2;
    }
    a = a_n;
    __builtin_prefetch((const void*)(A + arow + (unsigned)k + 512u), 0, 0);

    // Pin schedule per K step: all body loads first, then the WMMA group,
    // then the prefetch. With the 2-tile rotation every WMMA operand was
    // loaded >= 6 load-slots before the group ends -> partial waits only.
    __builtin_amdgcn_sched_group_barrier(0x020, 2u * NT + 2u, 0);  // VMEM reads
    __builtin_amdgcn_sched_group_barrier(0x008, NT, 0);            // WMMAs
    __builtin_amdgcn_sched_group_barrier(0x010, 1, 0);             // prefetch
  }

  // Epilogue: C/D layout -> lane holds (M = m0 + r + 8*hsel, N = t*16 + l15).
#pragma unroll
  for (int t = 0; t < NT; ++t) {
    int ng = t * 16 + l15;
    float bv = (bias != nullptr && ng < Nreal) ? bias[ng] : 0.0f;
#pragma unroll
    for (int r = 0; r < 8; ++r) {
      int m = m0 + r + 8 * hsel;
      float v = acc[t][r] + bv;
      if (relu) v = fmaxf(v, 0.0f);
      if (C != nullptr && ng < Nreal) C[(size_t)m * ldc + ng] = v;
      if (OutRM != nullptr) OutRM[(size_t)m * Npad + ng] = (_Float16)v;
      if (OutT != nullptr)
        OutT[(size_t)ng * M + m] = (_Float16)((ng < Nreal) ? v : 0.0f);
    }
  }
}

// ---- row softmax over 40 logits, one wave per row --------------------------
__global__ void softmax40_kernel(float* __restrict__ out, int M) {
  const int lane = threadIdx.x & 31;
  const int wave = threadIdx.x >> 5;
  const int row  = blockIdx.x * (blockDim.x >> 5) + wave;
  if (row >= M) return;
  float* p = out + (size_t)row * 40;
  float x0 = p[lane];
  float x1 = (lane < 8) ? p[lane + 32] : -3.4e38f;
  float mx = fmaxf(x0, x1);
#pragma unroll
  for (int off = 16; off > 0; off >>= 1) mx = fmaxf(mx, __shfl_xor(mx, off, 32));
  float e0 = __expf(x0 - mx);
  float e1 = (lane < 8) ? __expf(x1 - mx) : 0.0f;
  float s = e0 + e1;
#pragma unroll
  for (int off = 16; off > 0; off >>= 1) s += __shfl_xor(s, off, 32);
  float inv = 1.0f / s;
  p[lane] = e0 * inv;
  if (lane < 8) p[lane + 32] = e1 * inv;
}

// ---------------------------------------------------------------------------
extern "C" void kernel_launch(void* const* d_in, const int* in_sizes, int n_in,
                              void* d_out, int out_size, void* d_ws, size_t ws_size,
                              hipStream_t stream) {
  const float* x     = (const float*)d_in[0];   // [8192 x 512]
  const float* adj   = (const float*)d_in[1];   // [8192 x 8192]
  const float* w1    = (const float*)d_in[2];   // [512 x 128]
  const float* b1    = (const float*)d_in[3];   // [128]
  const float* w2    = (const float*)d_in[4];   // [128 x 40]
  const float* b2    = (const float*)d_in[5];   // [40]
  const float* alpha = (const float*)d_in[6];   // [1]
  float* out = (float*)d_out;                   // [8192 x 40]

  const int N = 8192, F = 512, H = 128, CP = 48, CR = 40;

  // Workspace carve-up (256B aligned regions).
  char* wp = (char*)d_ws;
  auto alloc = [&](size_t bytes) -> void* {
    void* p = (void*)wp;
    wp += (bytes + 255) & ~(size_t)255;
    return p;
  };
  _Float16* adjh = (_Float16*)alloc((size_t)N * N * 2);   // 134 MB
  _Float16* xh   = (_Float16*)alloc((size_t)N * F * 2);   // 8 MB
  _Float16* w1T  = (_Float16*)alloc((size_t)H * F * 2);   // [H x F]
  _Float16* w2T  = (_Float16*)alloc((size_t)CP * H * 2);  // [48 x H], pad rows 0
  _Float16* s1T  = (_Float16*)alloc((size_t)H * N * 2);   // [H x N]
  _Float16* hh   = (_Float16*)alloc((size_t)N * H * 2);   // [N x H] row-major
  _Float16* s2T  = (_Float16*)alloc((size_t)CP * N * 2);  // [48 x N]

  // P0: quantize adj -> fp16 (268 MB read, 134 MB write; the dominant traffic)
  {
    long n4 = (long)N * N / 4;
    quantize_adj_f16_kernel<<<(int)(n4 / 256), 256, 0, stream>>>(adj, alpha, adjh, n4);
  }
  // x -> fp16
  {
    long n4 = (long)N * F / 4;
    f32_to_f16_kernel<<<(int)(n4 / 256), 256, 0, stream>>>(x, xh, n4);
  }
  // w1^T, w2^T (padded to 48 rows)
  transpose_to_f16_kernel<<<(H * F + 255) / 256, 256, 0, stream>>>(w1, w1T, F, H, H);
  transpose_to_f16_kernel<<<(CP * H + 255) / 256, 256, 0, stream>>>(w2, w2T, H, CR, CP);

  const int WAVES = 4, BLK = WAVES * 32;
  const int gemm_grid = N / (16 * WAVES);  // 128 blocks, 16-row strip per wave

  // P1: s1T = (x @ w1)^T           M=N, K=512, Npad=128
  wmma_gemm_kernel<8, 512><<<gemm_grid, BLK, 0, stream>>>(
      xh, w1T, nullptr, nullptr, 0, nullptr, s1T, N, H, 0);

  // P2: h = relu(adj_q @ s1 + b1)  M=N, K=8192, Npad=128 (row-major f16 out)
  wmma_gemm_kernel<8, 8192><<<gemm_grid, BLK, 0, stream>>>(
      adjh, s1T, b1, nullptr, 0, hh, nullptr, N, H, 1);

  // P3: s2T = (h @ w2)^T           M=N, K=128, Npad=48 (pad rows zeroed)
  wmma_gemm_kernel<3, 128><<<gemm_grid, BLK, 0, stream>>>(
      hh, w2T, nullptr, nullptr, 0, nullptr, s2T, N, CR, 0);

  // P4: logits = adj_q @ s2 + b2 -> d_out fp32 [N x 40]
  wmma_gemm_kernel<3, 8192><<<gemm_grid, BLK, 0, stream>>>(
      adjh, s2T, b2, out, CR, nullptr, nullptr, N, CR, 0);

  // P5: softmax over 40 classes, one wave per row (8 rows / 256-thread block)
  softmax40_kernel<<<N / 8, 256, 0, stream>>>(out, N);
}
